// PoissonNLLLoss_37417755083225
// MI455X (gfx1250) — compile-verified, matrix operand order checked by reading, not run on hardware
//
#include <hip/hip_runtime.h>
#include <hip/hip_bf16.h>

// Problem constants (fixed by the reference's setup_inputs)
#define BB 16
#define HH 1024
#define WW 2048
#define HWPix (HH * WW)                     // 2^21 pixels per batch
#define NBINS 356
#define PIX_PER_BLOCK 16384
#define BLOCKS_PER_BATCH (HWPix / PIX_PER_BLOCK)   // 128
#define NBLOCKS (BB * BLOCKS_PER_BATCH)            // 2048
#define NTHREADS 256
#define VEC_PER_BLOCK (PIX_PER_BLOCK / 4)          // 4096 float4 per block
#define ITERS (VEC_PER_BLOCK / NTHREADS)           // 16

// ws layout (floats): cnt[B*NBINS] | sx[B*NBINS] | sy[B*NBINS] | expsum
#define WS_FLOATS (3 * BB * NBINS + 1)

typedef __attribute__((ext_vector_type(16))) _Float16 v16h;
typedef __attribute__((ext_vector_type(8)))  float    v8f;

union V16H { v16h v; _Float16 e[16]; };
union V8F  { v8f  v; float    e[8];  };

__global__ void pnll_zero_ws(float* __restrict__ ws) {
  int i = blockIdx.x * blockDim.x + threadIdx.x;
  if (i < WS_FLOATS) ws[i] = 0.0f;
}

__global__ __launch_bounds__(NTHREADS)
void pnll_stream(const float* __restrict__ logits,
                 const int*   __restrict__ label,
                 float*       __restrict__ ws) {
  __shared__ unsigned long long s_cnt_sx[NBINS];  // count<<40 | sum_x
  __shared__ unsigned int       s_sy[NBINS];
  __shared__ float              s_part[NTHREADS];

  const int tid = threadIdx.x;
  const int b   = blockIdx.x / BLOCKS_PER_BATCH;

  for (int i = tid; i < NBINS; i += NTHREADS) { s_cnt_sx[i] = 0ull; s_sy[i] = 0u; }
  __syncthreads();

  const float4* lg4 = (const float4*)logits;
  const int4*   lb4 = (const int4*)label;
  const long long blockVecBase = (long long)blockIdx.x * VEC_PER_BLOCK;
  const int localPixBase = (blockIdx.x % BLOCKS_PER_BATCH) * PIX_PER_BLOCK;

  float expAcc = 0.0f;
  #pragma unroll 4
  for (int it = 0; it < ITERS; ++it) {
    const int vlocal = it * NTHREADS + tid;          // 0..4095, coalesced
    const long long vg = blockVecBase + vlocal;
    __builtin_prefetch(lg4 + vg + 1024, 0, 0);       // -> global_prefetch_b8
    __builtin_prefetch(lb4 + vg + 1024, 0, 0);
    const float4 lg = lg4[vg];                       // global_load_b128
    const int4   lb = lb4[vg];                       // global_load_b128
    const int lp = localPixBase + vlocal * 4;        // batch-local pixel id
    const unsigned int y  = (unsigned)lp >> 11;      // W = 2048 = 2^11
    const unsigned int x0 = (unsigned)lp & 2047u;    // 4 pixels share a row

    expAcc += __expf(lg.x) + __expf(lg.y) + __expf(lg.z) + __expf(lg.w);

    // packed (count, sum_x): sum_x < 2^26, count < 2^15 -> no field carry
    atomicAdd(&s_cnt_sx[lb.x], (unsigned long long)(x0 + 0u) | (1ull << 40));
    atomicAdd(&s_cnt_sx[lb.y], (unsigned long long)(x0 + 1u) | (1ull << 40));
    atomicAdd(&s_cnt_sx[lb.z], (unsigned long long)(x0 + 2u) | (1ull << 40));
    atomicAdd(&s_cnt_sx[lb.w], (unsigned long long)(x0 + 3u) | (1ull << 40));
    atomicAdd(&s_sy[lb.x], y);
    atomicAdd(&s_sy[lb.y], y);
    atomicAdd(&s_sy[lb.z], y);
    atomicAdd(&s_sy[lb.w], y);
  }

  s_part[tid] = expAcc;
  __syncthreads();

  // Block-level exp reduction through the matrix pipe: with A = ones(16x32),
  // D[m][n] = sum_k B[k][n], so sum(D) = 16 * sum(B) for ANY element->VGPR
  // mapping. Each f32 partial is split into hi/lo f16 halves (f32-class
  // accuracy); 512 B-slots hold 256 hi + 256 lo values.
  if (tid < 32) {                                    // wave 0, EXEC all ones
    V16H a, bv; V8F c;
    #pragma unroll
    for (int j = 0; j < 16; ++j) a.e[j] = (_Float16)1.0f;
    #pragma unroll
    for (int j = 0; j < 8; ++j) {
      const float p = s_part[tid * 8 + j];
      const _Float16 hi = (_Float16)p;
      bv.e[j]     = hi;
      bv.e[j + 8] = (_Float16)(p - (float)hi);
    }
    #pragma unroll
    for (int j = 0; j < 8; ++j) c.e[j] = 0.0f;
    c.v = __builtin_amdgcn_wmma_f32_16x16x32_f16(
        /*neg_a=*/false, a.v, /*neg_b=*/false, bv.v,
        /*c_mod=*/(short)0, c.v, /*reuse_a=*/false, /*reuse_b=*/false);
    float t = 0.0f;
    #pragma unroll
    for (int j = 0; j < 8; ++j) t += c.e[j];
    #pragma unroll
    for (int off = 16; off >= 1; off >>= 1) t += __shfl_xor(t, off, 32);
    if (tid == 0) atomicAdd(&ws[3 * BB * NBINS], t * (1.0f / 16.0f));
  }

  // Flush privatized histogram (2048 blocks x 356 bins x 3 global atomics)
  float* g_cnt = ws;
  float* g_sx  = ws +     BB * NBINS;
  float* g_sy  = ws + 2 * BB * NBINS;
  for (int i = tid; i < NBINS; i += NTHREADS) {
    const unsigned long long v = s_cnt_sx[i];
    const unsigned int cnt = (unsigned int)(v >> 40);
    if (cnt) {
      const unsigned long long sx = v & ((1ull << 40) - 1ull);
      atomicAdd(&g_cnt[b * NBINS + i], (float)cnt);
      atomicAdd(&g_sx [b * NBINS + i], (float)sx);
      atomicAdd(&g_sy [b * NBINS + i], (float)s_sy[i]);
    }
  }
}

__global__ __launch_bounds__(NTHREADS)
void pnll_finalize(const float* __restrict__ logits,
                   const float* __restrict__ ws,
                   float*       __restrict__ out) {
  __shared__ float red[NTHREADS / 32];
  const float* g_cnt = ws;
  const float* g_sx  = ws +     BB * NBINS;
  const float* g_sy  = ws + 2 * BB * NBINS;

  float acc = 0.0f;
  for (int idx = threadIdx.x; idx < BB * NBINS; idx += NTHREADS) {
    const int id = idx % NBINS;
    const int b  = idx / NBINS;
    if (id > 100) {
      const float c = g_cnt[idx];
      if (c > 0.0f) {
        const float denom = fmaxf(c, 1.0f);
        const int cx = (int)(g_sx[idx] / denom);   // truncation, matches ref
        const int cy = (int)(g_sy[idx] / denom);
        acc += logits[(long long)b * HWPix + cy * WW + cx];
      }
    }
  }
  #pragma unroll
  for (int off = 16; off >= 1; off >>= 1) acc += __shfl_xor(acc, off, 32);
  if ((threadIdx.x & 31) == 0) red[threadIdx.x >> 5] = acc;
  __syncthreads();
  if (threadIdx.x == 0) {
    float inst = 0.0f;
    #pragma unroll
    for (int w = 0; w < NTHREADS / 32; ++w) inst += red[w];
    const float expsum = ws[3 * BB * NBINS];
    out[0] = expsum / (float)((long long)BB * HWPix) - inst / (float)BB;
  }
}

extern "C" void kernel_launch(void* const* d_in, const int* in_sizes, int n_in,
                              void* d_out, int out_size, void* d_ws, size_t ws_size,
                              hipStream_t stream) {
  (void)in_sizes; (void)n_in; (void)out_size; (void)ws_size;
  const float* logits = (const float*)d_in[0];
  const int*   label  = (const int*)d_in[1];
  float* out = (float*)d_out;
  float* ws  = (float*)d_ws;

  pnll_zero_ws<<<(WS_FLOATS + NTHREADS - 1) / NTHREADS, NTHREADS, 0, stream>>>(ws);
  pnll_stream<<<NBLOCKS, NTHREADS, 0, stream>>>(logits, label, ws);
  pnll_finalize<<<1, NTHREADS, 0, stream>>>(logits, ws, out);
}
